// SAGEConvolution_8718783611327
// MI455X (gfx1250) — matrix operand order repaired and checked by reading, not compiled, not run on hardware
//
#include <hip/hip_runtime.h>

#define F 128
#define EPSV 1e-6f

typedef __attribute__((ext_vector_type(16))) _Float16 v16h;
typedef __attribute__((ext_vector_type(8)))  float    v8f;

union FragU {
    v16h h;
    unsigned u[8];
};

// ---------------------------------------------------------------------------
// Kernel 0: zero the workspace (neigh_sum accumulator + degree accumulator)
// ---------------------------------------------------------------------------
__global__ void sage_zero_ws(float* __restrict__ p, long n) {
    long i = (long)blockIdx.x * blockDim.x + threadIdx.x;
    long stride = (long)gridDim.x * blockDim.x;
    for (; i < n; i += stride) p[i] = 0.0f;
}

// ---------------------------------------------------------------------------
// Kernel 1: edge scatter. One wave32 per edge: each lane moves float4
// (lane*16B within the 512B feature row -> fully coalesced), 4 f32 atomics
// into the L2-resident accumulator. Lane 0 bumps the degree counter.
// ---------------------------------------------------------------------------
__global__ __launch_bounds__(256)
void sage_scatter(const float* __restrict__ x,
                  const int*   __restrict__ row,
                  const int*   __restrict__ col,
                  float* __restrict__ neigh,
                  float* __restrict__ deg,
                  int E) {
    int gtid = blockIdx.x * blockDim.x + threadIdx.x;
    int e    = gtid >> 5;
    int lane = gtid & 31;
    if (e >= E) return;

    int s = col[e];   // source node
    int d = row[e];   // destination node

    const float4* src = (const float4*)(x + (long)s * F);
    float4 v = src[lane];

    float* dst = neigh + (long)d * F + lane * 4;
    atomicAdd(dst + 0, v.x);
    atomicAdd(dst + 1, v.y);
    atomicAdd(dst + 2, v.z);
    atomicAdd(dst + 3, v.w);

    if (lane == 0) atomicAdd(deg + d, 1.0f);
}

// ---------------------------------------------------------------------------
// Kernel 2: out = (neigh/deg) @ W^T + b  via v_wmma_f32_16x16x32_f16.
// Block = 256 threads (8 waves) handles 64 rows x 128 cols.
// Wave w owns output column tile [16w, 16w+16); iterates 4 row tiles.
// ---------------------------------------------------------------------------
__global__ __launch_bounds__(256)
void sage_gemm(const float* __restrict__ neigh,
               const float* __restrict__ deg,
               const float* __restrict__ W,     // [F_OUT][F_IN] row-major
               const float* __restrict__ bias,  // [F_OUT]
               float* __restrict__ out,         // [N][F_OUT]
               int N) {
    __shared__ __align__(16) _Float16 sW[F * F];    // W as f16, [n][k]
    __shared__ __align__(16) _Float16 sA[64 * F];   // mean slab as f16, [r][k]
    __shared__ float sB[F];

    const int tid = threadIdx.x;

    // Stage W -> f16 LDS
    for (int idx = tid; idx < F * F; idx += 256)
        sW[idx] = (_Float16)W[idx];
    if (tid < F) sB[tid] = bias[tid];

    const int rowbase = blockIdx.x * 64;

    // Stage mean slab -> f16 LDS (divide neighbor sum by degree + eps)
    for (int idx = tid; idx < 64 * F; idx += 256) {
        int r = idx >> 7;          // 0..63
        int k = idx & (F - 1);
        int grow = rowbase + r;
        float v = 0.0f;
        if (grow < N)
            v = neigh[(long)grow * F + k] / (deg[grow] + EPSV);
        sA[idx] = (_Float16)v;
    }
    __syncthreads();

    const int wave = tid >> 5;       // 0..7 -> column tile
    const int lane = tid & 31;
    const int g    = lane >> 4;      // lane group (0: lanes 0-15, 1: lanes 16-31)
    const int ln   = lane & 15;
    const int n    = wave * 16 + ln; // output column this lane contributes to (B/C layout)

    // Preload B fragments for all 4 K-steps.  B[k][n] = W[n][k]; this code's
    // dense 16-bit B 32x16 layout is: lane group g holds K in [16g,16g+16),
    // packed 2 halfs per VGPR -> 16 contiguous halfs = two b128 LDS loads.
    FragU bfr[4];
    #pragma unroll
    for (int s2 = 0; s2 < 4; ++s2) {
        const uint4* wsrc = (const uint4*)&sW[n * F + s2 * 32 + g * 16];
        uint4* bu = (uint4*)bfr[s2].u;
        bu[0] = wsrc[0];
        bu[1] = wsrc[1];
    }

    v8f acc[4] = {};

    #pragma unroll
    for (int t = 0; t < 4; ++t) {
        const int mrow = t * 16 + ln;  // A-matrix row = lane % 16 (both groups)
        #pragma unroll
        for (int s2 = 0; s2 < 4; ++s2) {
            // A 16x32 f16 layout (matching B): lane group g holds K in
            // [16g,16g+16), 16 contiguous halfs (16B-aligned) -> two b128 LDS loads.
            FragU af;
            uint4* au = (uint4*)af.u;
            au[0] = *(const uint4*)&sA[mrow * F + s2 * 32 + g * 16];
            au[1] = *(const uint4*)&sA[mrow * F + s2 * 32 + g * 16 + 8];

            acc[t] = __builtin_amdgcn_wmma_f32_16x16x32_f16(
                /*neg_a=*/false, af.h,
                /*neg_b=*/false, bfr[s2].h,
                /*c_mod=*/(short)0, acc[t],
                /*reuse_a=*/false, /*reuse_b=*/false);
        }
    }

    // Epilogue: C/D layout -> VGPR i holds row M = 8g + i (within 16-row tile),
    // column N = lane % 16.  Add bias, store.
    const float bv = sB[n];
    #pragma unroll
    for (int t = 0; t < 4; ++t) {
        #pragma unroll
        for (int i = 0; i < 8; ++i) {
            int grow = rowbase + t * 16 + g * 8 + i;
            if (grow < N)
                out[(long)grow * F + n] = acc[t][i] + bv;
        }
    }
}

// ---------------------------------------------------------------------------
extern "C" void kernel_launch(void* const* d_in, const int* in_sizes, int n_in,
                              void* d_out, int out_size, void* d_ws, size_t ws_size,
                              hipStream_t stream) {
    const float* x   = (const float*)d_in[0];
    const int*   row = (const int*)  d_in[1];
    const int*   col = (const int*)  d_in[2];
    const float* W   = (const float*)d_in[3];
    const float* b   = (const float*)d_in[4];
    float* out = (float*)d_out;

    const int N = in_sizes[0] / F;   // 40000
    const int E = in_sizes[1];       // 640000

    // Workspace: [N*F] neighbor-sum accumulator, then [N] degree accumulator.
    float* neigh = (float*)d_ws;
    float* deg   = neigh + (long)N * F;

    // 1) Zero accumulators (harness poisons ws with 0xAA).
    long zn = (long)N * F + N;
    sage_zero_ws<<<2048, 256, 0, stream>>>(neigh, zn);

    // 2) Edge scatter: one wave per edge.
    long scatter_threads = (long)E * 32;
    int scatter_blocks = (int)((scatter_threads + 255) / 256);
    sage_scatter<<<scatter_blocks, 256, 0, stream>>>(x, row, col, neigh, deg, E);

    // 3) WMMA GEMM with fused mean + bias.
    int gemm_blocks = (N + 63) / 64;
    sage_gemm<<<gemm_blocks, 256, 0, stream>>>(neigh, deg, W, b, out, N);
}
